// VRAE_RNN_61899068670614
// MI455X (gfx1250) — matrix operand order repaired and checked by reading, not compiled, not run on hardware
//
#include <hip/hip_runtime.h>
#include <math.h>

// Problem dims (VRAE): B=128, T=1024, F=32, H=256
constexpr int NBv = 128;
constexpr int NTv = 1024;
constexpr int NFv = 32;
constexpr int NHv = 256;

typedef __attribute__((ext_vector_type(16))) __bf16 bf16x16;
typedef __attribute__((ext_vector_type(8)))  float  f32x8;
typedef __attribute__((ext_vector_type(4)))  unsigned int u32x4;

union Frag {
    bf16x16 v;
    u32x4   q[2];
    unsigned int u[8];
};

// f32 -> bf16 by truncation (1 VALU op)
__device__ __forceinline__ unsigned short f2bf(float f) {
    return (unsigned short)(__float_as_uint(f) >> 16);
}
// pack two f32 -> packed bf16 pair with a single v_perm_b32:
// D = { hi[31:24], hi[23:16], lo[31:24], lo[23:16] }
__device__ __forceinline__ unsigned pack2(float lo, float hi) {
    return __builtin_amdgcn_perm(__float_as_uint(hi), __float_as_uint(lo), 0x07060302u);
}

// branch-free tanh: clamp + 1 - 2/(e^{2x}+1)  (v_exp + v_rcp, no EXEC divergence)
__device__ __forceinline__ float fast_tanh(float v) {
    v = fminf(9.0f, fmaxf(-9.0f, v));
    float e = __expf(2.0f * v);
    float r = __builtin_amdgcn_rcpf(e + 1.0f);
    return fmaf(-2.0f, r, 1.0f);
}

__device__ __forceinline__ f32x8 wmma_bf16(bf16x16 a, bf16x16 b, f32x8 c) {
    // D = A(16x32 bf16) x B(32x16 bf16) + C(16x16 f32)
    return __builtin_amdgcn_wmma_f32_16x16x32_bf16(false, a, false, b, (short)0, c, false, false);
}

// B-fragment (K=32 x N=16, bf16) for D = A*B where B[k][n] = W[n][k];
// W is row-major f32 with leading dim ldW. contiguous k order (A frags use the same):
//   lanes 0-15: n = lane,    K = kb+0..15 (reg v holds K=kb+2v, kb+2v+1)
//   lanes16-31: n = lane-16, K = kb+16..31
__device__ __forceinline__ bf16x16 load_bfrag(const float* W, int ldW, int n0, int kb, int lane) {
    const float* p = W + (size_t)(n0 + (lane & 15)) * ldW + kb + ((lane & 16) ? 16 : 0);
    Frag f;
#pragma unroll
    for (int v = 0; v < 8; ++v) f.u[v] = pack2(p[2 * v], p[2 * v + 1]);
    return f.v;
}

// Same, but B[k][n] = Wa[n][k] + Wb[n][k]  (decoder: Wih + Whh since y==z after step 1)
__device__ __forceinline__ bf16x16 load_bfrag_sum(const float* Wa, const float* Wb, int ldW,
                                                  int n0, int kb, int lane) {
    size_t off = (size_t)(n0 + (lane & 15)) * ldW + kb + ((lane & 16) ? 16 : 0);
    const float* pa = Wa + off;
    const float* pb = Wb + off;
    Frag f;
#pragma unroll
    for (int v = 0; v < 8; ++v)
        f.u[v] = pack2(pa[2 * v] + pb[2 * v], pa[2 * v + 1] + pb[2 * v + 1]);
    return f.v;
}

// A-fragment (M=16 x K=32 bf16) from LDS state [16][NHv] bf16. contiguous k order (same as B frags):
//   lane L: m = L&15; k-base = kb + (L<16 ? 0 : 16)
//   regs v0..3 hold K = kbase+0..7 (contiguous 16B), v4..7 hold K = kbase+8..15
__device__ __forceinline__ bf16x16 load_afrag(const unsigned short* hb, int kb, int lane) {
    const unsigned short* p = hb + (lane & 15) * NHv + kb + ((lane & 16) ? 16 : 0);
    Frag f;
    f.q[0] = *(const u32x4*)(p);
    f.q[1] = *(const u32x4*)(p + 8);
    return f.v;
}

// Software-pipelined 8-chunk K-accumulation: A from LDS state, resident B frags.
// Loads for chunk c+1 are issued before the WMMA of chunk c retires.
__device__ __forceinline__ f32x8 mm_chain(const unsigned short* hb, const bf16x16* bw,
                                          int lane, f32x8 acc) {
    bf16x16 a = load_afrag(hb, 0, lane);
#pragma unroll
    for (int c = 0; c < 8; ++c) {
        bf16x16 an;
        if (c < 7) an = load_afrag(hb, (c + 1) * 32, lane);
        acc = wmma_bf16(a, bw[c], acc);
        if (c < 7) a = an;
    }
    return acc;
}

// A-fragment from x[b, t, 0..31] (global f32, K = F = 32)
__device__ __forceinline__ bf16x16 load_afrag_x(const float* x, int m0, int t, int lane) {
    const float* p = x + ((size_t)(m0 + (lane & 15)) * NTv + t) * NFv;
    int fb = (lane & 16) ? 16 : 0;
    Frag f;
#pragma unroll
    for (int v = 0; v < 4; ++v) f.u[v] = pack2(p[fb + 2 * v], p[fb + 2 * v + 1]);
#pragma unroll
    for (int v = 0; v < 4; ++v) f.u[4 + v] = pack2(p[fb + 8 + 2 * v], p[fb + 9 + 2 * v]);
    return f.v;
}

// C-tile (16x16 f32): reg r -> row (mb+r), lane -> col n. Apply bias+tanh, store bf16 state.
__device__ __forceinline__ void store_state_tanh(unsigned short* hb, f32x8 acc, float bias,
                                                 int n0, int lane) {
    int n  = n0 + (lane & 15);
    int mb = (lane & 16) ? 8 : 0;
#pragma unroll
    for (int r = 0; r < 8; ++r)
        hb[(mb + r) * NHv + n] = f2bf(fast_tanh(acc[r] + bias));
}

__global__ __launch_bounds__(512, 1)
void vrae_fused(const float* __restrict__ x,    const float* __restrict__ eps,
                const float* __restrict__ eWih, const float* __restrict__ eWhh,
                const float* __restrict__ eb,
                const float* __restrict__ dWih, const float* __restrict__ dWhh,
                const float* __restrict__ db,
                const float* __restrict__ Wmu,  const float* __restrict__ bmu,
                const float* __restrict__ Wlv,  const float* __restrict__ blv,
                const float* __restrict__ Wmx,  const float* __restrict__ bmx,
                const float* __restrict__ Wlx,  const float* __restrict__ blx,
                float* __restrict__ out)
{
    __shared__ unsigned short hbuf[2][16 * NHv];   // bf16 state ping-pong, 16 KB

    const int lane = threadIdx.x & 31;
    const int wave = threadIdx.x >> 5;     // 16 waves
    const int m0   = blockIdx.x * 16;      // batch tile (8 blocks)
    const int n0   = wave * 16;            // this wave's N-tile of H
    const int nl   = lane & 15;
    const int mb   = (lane & 16) ? 8 : 0;

    // h0 = 0
    for (int i = threadIdx.x; i < 16 * NHv; i += blockDim.x) hbuf[0][i] = 0;

    // ---- encoder weights resident in VGPRs (reused 1024x) ----
    bf16x16 bX = load_bfrag(eWih, NFv, n0, 0, lane);     // input projection (K=F=32)
    bf16x16 bW[8];
#pragma unroll
    for (int c = 0; c < 8; ++c) bW[c] = load_bfrag(eWhh, NHv, n0, c * 32, lane);
    float bias = eb[n0 + nl];
    __syncthreads();

    // ---- encoder scan: h = tanh(x_t Wih^T + h Whh^T + b) ----
    int p = 0;
    for (int t = 0; t < NTv; ++t) {
        if (t + 1 < NTv)
            __builtin_prefetch(x + ((size_t)(m0 + nl) * NTv + t + 1) * NFv, 0, 1);
        f32x8 acc = {};
        acc = wmma_bf16(load_afrag_x(x, m0, t, lane), bX, acc);
        acc = mm_chain(hbuf[p], bW, lane, acc);
        store_state_tanh(hbuf[p ^ 1], acc, bias, n0, lane);
        p ^= 1;
        __syncthreads();
    }

    // ---- posterior heads + reparameterization: z0 = eps*exp(lv/2) + mu ----
    {
        f32x8 am = {}, al = {};
#pragma unroll
        for (int c = 0; c < 8; ++c) {
            bf16x16 ah = load_afrag(hbuf[p], c * 32, lane);
            am = wmma_bf16(ah, load_bfrag(Wmu, NHv, n0, c * 32, lane), am);
            al = wmma_bf16(ah, load_bfrag(Wlv, NHv, n0, c * 32, lane), al);
        }
        float bm = bmu[n0 + nl], bl = blv[n0 + nl];
#pragma unroll
        for (int r = 0; r < 8; ++r) {
            int b = m0 + mb + r;
            int n = n0 + nl;
            float muv = am[r] + bm;
            float lvv = al[r] + bl;
            out[(size_t)b * NHv + n] = muv;                                // mu_post
            out[(size_t)(NBv * NHv) + (size_t)b * NHv + n] = lvv;          // logsigma2_post
            float z0 = eps[(size_t)b * NHv + n] * __expf(0.5f * lvv) + muv;
            hbuf[p ^ 1][(mb + r) * NHv + n] = f2bf(z0);                    // decoder init state
        }
    }
    __syncthreads();
    p ^= 1;   // z0 now in hbuf[p]

    // ---- decoder weights: after step 1, input == hidden, so W = Wih + Whh ----
#pragma unroll
    for (int c = 0; c < 8; ++c) bW[c] = load_bfrag_sum(dWih, dWhh, NHv, n0, c * 32, lane);
    bias = db[n0 + nl];

    // fused reconstruction heads: waves 0..3 each own (head, f-tile); N=F=32 per head
    const bool  projw = (wave < 4);
    const int   head  = (wave >> 1) & 1;            // 0 -> mu_xhat, 1 -> logsigma2_xhat
    const int   fb2   = (wave & 1) * 16;
    const float* Wp   = head ? Wlx : Wmx;
    const float pbias = (head ? blx : bmx)[fb2 + nl];
    const size_t obase = (size_t)(2 * NBv * NHv) + (size_t)head * NBv * NTv * NFv;
    bf16x16 bP[8];
    if (projw) {
#pragma unroll
        for (int c = 0; c < 8; ++c) bP[c] = load_bfrag(Wp, NHv, fb2, c * 32, lane);
    }
    __syncthreads();

    // ---- decoder scan + fused output projections ----
    for (int t = 1; t <= NTv; ++t) {
        f32x8 acc = {};
        if (t == 1) {
            // y1 = tanh(z0 Whh^T + b)  (y0 == 0 kills the Wih term)
#pragma unroll
            for (int c = 0; c < 8; ++c)
                acc = wmma_bf16(load_afrag(hbuf[p], c * 32, lane),
                                load_bfrag(dWhh, NHv, n0, c * 32, lane), acc);
        } else {
            acc = mm_chain(hbuf[p], bW, lane, acc);
        }
        store_state_tanh(hbuf[p ^ 1], acc, bias, n0, lane);
        p ^= 1;
        __syncthreads();
        if (projw) {   // wave-uniform branch: EXEC stays all-ones inside
            f32x8 pa = {};
            pa = mm_chain(hbuf[p], bP, lane, pa);
#pragma unroll
            for (int r = 0; r < 8; ++r) {
                int b = m0 + mb + r;
                out[obase + ((size_t)b * NTv + (t - 1)) * NFv + fb2 + nl] = pa[r] + pbias;
            }
        }
    }
}

extern "C" void kernel_launch(void* const* d_in, const int* in_sizes, int n_in,
                              void* d_out, int out_size, void* d_ws, size_t ws_size,
                              hipStream_t stream) {
    const float* x    = (const float*)d_in[0];
    const float* eps  = (const float*)d_in[1];
    const float* eWih = (const float*)d_in[2];
    const float* eWhh = (const float*)d_in[3];
    const float* eb   = (const float*)d_in[4];
    const float* dWih = (const float*)d_in[5];
    const float* dWhh = (const float*)d_in[6];
    const float* db   = (const float*)d_in[7];
    const float* Wmu  = (const float*)d_in[8];
    const float* bmu  = (const float*)d_in[9];
    const float* Wlv  = (const float*)d_in[10];
    const float* blv  = (const float*)d_in[11];
    const float* Wmx  = (const float*)d_in[12];
    const float* bmx  = (const float*)d_in[13];
    const float* Wlx  = (const float*)d_in[14];
    const float* blx  = (const float*)d_in[15];
    float* out = (float*)d_out;

    dim3 grid(NBv / 16);   // 8 persistent workgroups, one 16-row batch tile each
    dim3 block(512);       // 16 waves: one H n-tile per wave
    hipLaunchKernelGGL(vrae_fused, grid, block, 0, stream,
                       x, eps, eWih, eWhh, eb, dWih, dWhh, db,
                       Wmu, bmu, Wlv, blv, Wmx, bmx, Wlx, blx, out);
}
